// ClockworkRNN_13795434954913
// MI455X (gfx1250) — matrix operand order repaired and checked by reading, not compiled, not run on hardware
//
#include <hip/hip_runtime.h>
#include <math.h>

// ---------------------------------------------------------------------------
// Clockwork RNN for MI455X (gfx1250, wave32, WMMA f32<-f16 16x16x32)
//   Phase 0: X (b,t,k) f32  ->  X16 (t*64+b, k) f16   (scan-order, contiguous
//            A tiles for the GEMM; halves GEMM A read traffic)
//   Phase 1: Xw[t][b][o] = X16 @ W + b   (M=64 blocking -> Wt L2 traffic /4,
//            A tile staged to LDS via global_load_async_to_lds_b128/ASYNCcnt)
//   Phase 2: sequential scan on one WGP, h resident in 64KB LDS (f16),
//            32 waves (1 n-tile each), v_tanh, two barriers per step.
// ---------------------------------------------------------------------------

typedef __attribute__((ext_vector_type(16))) _Float16 v16h;
typedef __attribute__((ext_vector_type(8)))  _Float16 h8;
typedef __attribute__((ext_vector_type(4)))  _Float16 h4;
typedef __attribute__((ext_vector_type(8)))  float    v8f;

#define T_STEPS 2048
#define BATCH   64
#define DIN     256
#define DOUT    512

union HU { v16h v; h8 h[2]; };

__device__ __forceinline__ v8f wmma_f16(v16h a, v16h b, v8f c) {
  return __builtin_amdgcn_wmma_f32_16x16x32_f16(
      false, a, false, b, (short)0, c, false, false);
}

__device__ __forceinline__ float fast_tanh(float x) {
#if __has_builtin(__builtin_amdgcn_tanhf)
  return __builtin_amdgcn_tanhf(x);          // v_tanh_f32
#else
  const float e = __expf(2.0f * x);          // v_exp_f32 path
  return 1.0f - 2.0f * __builtin_amdgcn_rcpf(e + 1.0f);
#endif
}

// Async copy of 16 bytes global -> LDS (GV mode, tracked by ASYNCcnt).
// vdst = wave-relative LDS byte address (low 32 bits of generic shared ptr).
__device__ __forceinline__ void copy16_g2l(const _Float16* g, _Float16* l) {
  const unsigned lds          = (unsigned)(uintptr_t)l;
  const unsigned long long ga = (unsigned long long)(uintptr_t)g;
  asm volatile("global_load_async_to_lds_b128 %0, %1, off"
               :: "v"(lds), "v"(ga) : "memory");
}

__device__ __forceinline__ void wait_async_copies() {
  asm volatile("s_wait_asynccnt 0x0" ::: "memory");
}

// X (b,t,k) f32 -> X16 (t*64+b, k) f16; 4 elements per thread, both sides coalesced.
__global__ void __launch_bounds__(256) k_convert_X(
    const float* __restrict__ X, _Float16* __restrict__ X16) {
  const size_t o = ((size_t)blockIdx.x * 256 + threadIdx.x) * 4;
  const int k   = (int)(o & (DIN - 1));
  const int row = (int)(o >> 8);          // t*64 + b
  const int t = row >> 6;
  const int b = row & 63;
  const float4 v = *(const float4*)(X + ((size_t)b * T_STEPS + t) * DIN + k);
  h4 hv;
  hv[0] = (_Float16)v.x; hv[1] = (_Float16)v.y;
  hv[2] = (_Float16)v.z; hv[3] = (_Float16)v.w;
  *(h4*)(X16 + o) = hv;
}

// dst[C][R] (f16) = transpose of src[R][C] (f32); coalesced writes.
__global__ void k_transpose_f16(const float* __restrict__ src,
                                _Float16* __restrict__ dst, int R, int C) {
  int idx = blockIdx.x * 256 + threadIdx.x;
  if (idx < R * C) {
    int r = idx % R;
    int c = idx / R;
    dst[idx] = (_Float16)src[(size_t)r * C + c];  // idx == c*R + r
  }
}

// One block: 64 output rows x full N=512. 8 waves; wave w owns n-tiles 4w..4w+3,
// all 4 m-tiles (B fragment reused across the 4 m-tiles per issue group).
__global__ void __launch_bounds__(256) cwrnn_gemm(
    const _Float16* __restrict__ X16, const _Float16* __restrict__ Wt,
    const float* __restrict__ bias, float* __restrict__ Xw) {
  __shared__ _Float16 As[64 * DIN];  // 32 KB contiguous A tile (f16)

  const int tid  = threadIdx.x;
  const int wave = tid >> 5;
  const int lane = tid & 31;
  const int lm   = lane & 15;
  const int hi   = lane >> 4;
  const int rowbase = blockIdx.x * 64;   // global row r = t*64 + b

  // Stage contiguous 32 KB A tile: 256 threads x 8 x 16B, async to LDS.
  {
    const _Float16* src = X16 + (size_t)rowbase * DIN;
    #pragma unroll
    for (int j = 0; j < 8; ++j) {
      const int off = j * 2048 + tid * 8;     // in f16 elements (16B granules)
      copy16_g2l(src + off, As + off);
    }
    wait_async_copies();
  }
  __syncthreads();

  v8f acc[4][4];
  #pragma unroll
  for (int mt = 0; mt < 4; ++mt)
    #pragma unroll
    for (int j = 0; j < 4; ++j)
      #pragma unroll
      for (int v = 0; v < 8; ++v) acc[mt][j][v] = 0.0f;

  for (int k0 = 0; k0 < DIN; k0 += 32) {
    HU a[4];
    #pragma unroll
    for (int mt = 0; mt < 4; ++mt) {
      const _Float16* ap = As + (mt * 16 + lm) * DIN;
      a[mt].h[0] = *(const h8*)(ap + k0 + hi * 8);
      a[mt].h[1] = *(const h8*)(ap + k0 + 16 + hi * 8);
    }
    #pragma unroll
    for (int j = 0; j < 4; ++j) {
      const int n = wave * 64 + j * 16 + lm;
      const _Float16* bp = Wt + (size_t)n * DIN + k0 + hi * 8;
      HU bf;
      bf.h[0] = *(const h8*)(bp);
      bf.h[1] = *(const h8*)(bp + 16);
      #pragma unroll
      for (int mt = 0; mt < 4; ++mt)
        acc[mt][j] = wmma_f16(a[mt].v, bf.v, acc[mt][j]);
    }
  }

  #pragma unroll
  for (int j = 0; j < 4; ++j) {
    const int n = wave * 64 + j * 16 + lm;
    const float bv = bias[n];
    #pragma unroll
    for (int mt = 0; mt < 4; ++mt)
      #pragma unroll
      for (int v = 0; v < 8; ++v) {
        const int m = mt * 16 + v + 8 * hi;
        Xw[(size_t)(rowbase + m) * DOUT + n] = acc[mt][j][v] + bv;
      }
  }
}

// Single workgroup, 32 waves. h (64x512 f16, post-tanh) lives in LDS.
// Wave w owns n-tile w (clockwork block w/4) x all 4 m-tiles.
__global__ void __launch_bounds__(1024) cwrnn_scan(
    const float* __restrict__ Xw, const _Float16* __restrict__ Wct,
    float* __restrict__ out) {
  __shared__ _Float16 hbuf[BATCH * DOUT];  // 64 KB

  const int tid  = threadIdx.x;
  const int wave = tid >> 5;               // 0..31 == n-tile
  const int lane = tid & 31;
  const int lm   = lane & 15;
  const int hi   = lane >> 4;

  for (int e = tid; e < BATCH * DOUT; e += 1024) hbuf[e] = (_Float16)0.0f;
  __syncthreads();

  const int nt  = wave;
  const int i   = nt >> 2;                 // clockwork block 0..7
  const int kb  = i * 64;
  const int Kc  = DOUT - 64 * i;
  const int pm  = (1 << i) - 1;            // period mask
  const int col = nt * 16 + lm;
  const _Float16* wb =
      Wct + (size_t)2048 * i * (17 - i) + (size_t)((nt & 3) * 16 + lm) * Kc;

  for (int t = 0; t < T_STEPS; ++t) {
    if (t + 1 < T_STEPS)  // pull next step's x_t slab toward the caches
      __builtin_prefetch(Xw + (size_t)(t + 1) * BATCH * DOUT + tid * 32, 0, 1);

    v8f res[4];
    const bool active = (t & pm) == 0;
    #pragma unroll
    for (int mt = 0; mt < 4; ++mt) {
      v8f c;
      const float* xp =
          Xw + ((size_t)t * BATCH + mt * 16 + 8 * hi) * DOUT + col;
      #pragma unroll
      for (int v = 0; v < 8; ++v) c[v] = xp[(size_t)v * DOUT];

      if (active) {
        const _Float16* ap = hbuf + (mt * 16 + lm) * DOUT;
        for (int k0 = kb; k0 < DOUT; k0 += 32) {
          HU a, bf;
          a.h[0] = *(const h8*)(ap + k0 + hi * 8);
          a.h[1] = *(const h8*)(ap + k0 + 16 + hi * 8);
          const _Float16* bp = wb + (k0 - kb) + hi * 8;
          bf.h[0] = *(const h8*)(bp);
          bf.h[1] = *(const h8*)(bp + 16);
          c = wmma_f16(a.v, bf.v, c);
        }
      } else {
        // reference re-applies tanh to the held state when block is inactive
        #pragma unroll
        for (int v = 0; v < 8; ++v)
          c[v] = (float)hbuf[(mt * 16 + v + 8 * hi) * DOUT + col];
      }
      #pragma unroll
      for (int v = 0; v < 8; ++v) c[v] = fast_tanh(c[v]);
      res[mt] = c;
    }
    __syncthreads();  // all reads of h done

    #pragma unroll
    for (int mt = 0; mt < 4; ++mt)
      #pragma unroll
      for (int v = 0; v < 8; ++v) {
        const int row = mt * 16 + v + 8 * hi;
        hbuf[row * DOUT + col] = (_Float16)res[mt][v];
        if (t == T_STEPS - 1) out[(size_t)row * DOUT + col] = res[mt][v];
      }
    __syncthreads();  // writes visible before next step's reads
  }
}

extern "C" void kernel_launch(void* const* d_in, const int* in_sizes, int n_in,
                              void* d_out, int out_size, void* d_ws,
                              size_t ws_size, hipStream_t stream) {
  const float* X    = (const float*)d_in[0];
  const float* W    = (const float*)d_in[1];
  const float* bias = (const float*)d_in[2];

  char* ws = (char*)d_ws;
  float* Xw = (float*)ws;                                   // 256 MiB f32
  size_t off = (size_t)T_STEPS * BATCH * DOUT * sizeof(float);
  _Float16* X16 = (_Float16*)(ws + off);                    // 64 MiB f16
  off += (size_t)T_STEPS * BATCH * DIN * sizeof(_Float16);
  _Float16* Wt = (_Float16*)(ws + off);                     // 512x256 f16
  off += (size_t)DOUT * DIN * sizeof(_Float16);
  _Float16* Wct = (_Float16*)(ws + off);                    // 288 KiB f16

  // X -> X16 (scan-order rows, f16)
  {
    const size_t nthreads = (size_t)T_STEPS * BATCH * DIN / 4;
    k_convert_X<<<(unsigned)(nthreads / 256), 256, 0, stream>>>(X, X16);
  }
  // W (256x512) -> Wt (512x256) f16
  {
    int nel = DIN * DOUT;
    k_transpose_f16<<<(nel + 255) / 256, 256, 0, stream>>>(W, Wt, DIN, DOUT);
  }
  // Wc_i (Kc x 64) -> Wct_i (64 x Kc) f16, packed back-to-back
  size_t o = 0;
  for (int i = 0; i < 8; ++i) {
    const float* Wc = (const float*)d_in[3 + i];
    const int Kc = DOUT - 64 * i;
    const int nel = Kc * 64;
    k_transpose_f16<<<(nel + 255) / 256, 256, 0, stream>>>(Wc, Wct + o, Kc, 64);
    o += (size_t)nel;  // matches 2048*i*(17-i) cumulative formula
  }

  cwrnn_gemm<<<(T_STEPS * BATCH) / 64, 256, 0, stream>>>(X16, Wt, bias, Xw);
  cwrnn_scan<<<1, 1024, 0, stream>>>(Xw, Wct, (float*)d_out);
}
